// DecoderBlock_1030792151498
// MI455X (gfx1250) — compile-verified, matrix operand order checked by reading, not run on hardware
//
#include <hip/hip_runtime.h>
#include <hip/hip_bf16.h>

typedef __attribute__((ext_vector_type(16))) _Float16 v16h;
typedef __attribute__((ext_vector_type(8)))  float    v8f;

constexpr int NTOK = 16 * 1024;   // N*L
constexpr int HD   = 512;         // C*K
constexpr int CCH  = 8;           // channels
constexpr int KD   = 64;          // per-channel dim
constexpr int MD   = 2048;        // mlp dim

// ---------------------------------------------------------------------------
// WMMA helpers (wave32, 16x16x32 f16 -> f32)
// A-frag: lane holds row M=lane%16; elements cover K = k0 + {lo..lo+7, 16+lo..16+lo+7},
// lo = (lane>=16)?8:0, packed 2 halves/VGPR  -> two 16B loads.
// B is stored transposed in memory ([N][K]) so the same loader works with row = column N.
// D: vgpr r, lane l -> (M = r + (l>=16?8:0), N = l%16).
// ---------------------------------------------------------------------------
union V16U { v16h h; int4 q[2]; };

__device__ __forceinline__ v16h load16(const _Float16* row, int k0, int lane) {
  const int lo = (lane & 16) ? 8 : 0;
  V16U u;
  u.q[0] = *(const int4*)(row + k0 + lo);
  u.q[1] = *(const int4*)(row + k0 + lo + 16);
  return u.h;
}

__device__ __forceinline__ v8f wmma16(v16h a, v16h b, v8f c) {
  return __builtin_amdgcn_wmma_f32_16x16x32_f16(false, a, false, b, (short)0, c, false, false);
}

// ---------------------------------------------------------------------------
// Weight prep: dst[b][n][r] = (f16) src[b][r][n]   (transpose + convert)
// ---------------------------------------------------------------------------
__global__ void __launch_bounds__(256) transpose_f16_kernel(
    const float* __restrict__ src, _Float16* __restrict__ dst,
    int B, int R, int N) {
  int idx = blockIdx.x * 256 + threadIdx.x;
  int total = B * R * N;
  if (idx >= total) return;
  int r = idx % R;
  int n = (idx / R) % N;
  int b = idx / (R * N);
  dst[idx] = (_Float16)src[((size_t)b * R + r) * N + n];
}

// ---------------------------------------------------------------------------
// LN1: one wave per token (8 tokens / 256-thread block)
// ---------------------------------------------------------------------------
__global__ void __launch_bounds__(256) ln1_kernel(
    const float* __restrict__ x, const float* __restrict__ w,
    const float* __restrict__ b, float* __restrict__ of32,
    _Float16* __restrict__ of16) {
  int wave = threadIdx.x >> 5, lane = threadIdx.x & 31;
  int token = blockIdx.x * 8 + wave;
  const float* row = x + (size_t)token * HD;
  float v[16]; float s = 0.f;
#pragma unroll
  for (int i = 0; i < 16; i++) { v[i] = row[lane + 32 * i]; s += v[i]; }
#pragma unroll
  for (int off = 16; off > 0; off >>= 1) s += __shfl_xor(s, off, 32);
  float mu = s * (1.f / 512.f);
  float var = 0.f;
#pragma unroll
  for (int i = 0; i < 16; i++) { float d = v[i] - mu; var += d * d; }
#pragma unroll
  for (int off = 16; off > 0; off >>= 1) var += __shfl_xor(var, off, 32);
  float rstd = rsqrtf(var * (1.f / 512.f) + 1e-6f);
#pragma unroll
  for (int i = 0; i < 16; i++) {
    int e = lane + 32 * i;
    float o = (v[i] - mu) * rstd * w[e] + b[e];
    of32[(size_t)token * HD + e] = o;
    of16[(size_t)token * HD + e] = (_Float16)o;
  }
}

// ---------------------------------------------------------------------------
// Criss-cross core: ctx[m] = sum_c v[c] * softmax_m(q[c]*k[m]/8)
// ---------------------------------------------------------------------------
__device__ __forceinline__ void cc_attn8(const float* qv, const float* kk,
                                         const float* vv, float* cacc) {
#pragma unroll
  for (int m = 0; m < 8; m++) cacc[m] = 0.f;
#pragma unroll
  for (int ci = 0; ci < 8; ci++) {
    float e[8], Z = 0.f;
#pragma unroll
    for (int m = 0; m < 8; m++) { e[m] = __expf(qv[ci] * kk[m] * 0.125f); Z += e[m]; }
    float wgt = vv[ci] / Z;
#pragma unroll
    for (int m = 0; m < 8; m++) cacc[m] += wgt * e[m];
  }
}

// ---------------------------------------------------------------------------
// Self-attention block: 16 tokens / block, 8 waves (wave == channel for GEMMs)
// qkv WMMA -> attn -> fc_s WMMA -> residual -> xin2
// ---------------------------------------------------------------------------
__global__ void __launch_bounds__(256) self_attn_kernel(
    const _Float16* __restrict__ xin16, const float* __restrict__ xin_f32,
    const _Float16* __restrict__ wT_qkv, const float* __restrict__ b_qkv,
    const _Float16* __restrict__ wT_fcs, const float* __restrict__ b_fcs,
    float* __restrict__ xin2) {
  __shared__ alignas(16) _Float16 qkv16[16][CCH][192];  // 48 KB
  __shared__ alignas(16) _Float16 sa16[16][CCH][KD];    // 16 KB
  int tid = threadIdx.x, wave = tid >> 5, lane = tid & 31;
  int tok0 = blockIdx.x * 16;

  // ---- qkv GEMM: [16,64] x [64,192] per channel ----
  {
    int c = wave;
    const _Float16* Arow = xin16 + (size_t)(tok0 + (lane & 15)) * HD + c * KD;
    v16h a0 = load16(Arow, 0, lane), a1 = load16(Arow, 32, lane);
#pragma unroll
    for (int nt = 0; nt < 12; nt++) {
      int n = nt * 16 + (lane & 15);
      const _Float16* Brow = wT_qkv + ((size_t)c * 192 + n) * KD;
      v16h b0 = load16(Brow, 0, lane), b1 = load16(Brow, 32, lane);
      v8f acc = {};
      acc = wmma16(a0, b0, acc);
      acc = wmma16(a1, b1, acc);
      float bias = b_qkv[c * 192 + n];
      int mhi = (lane & 16) ? 8 : 0;
#pragma unroll
      for (int r = 0; r < 8; r++) qkv16[r + mhi][c][n] = (_Float16)(acc[r] + bias);
    }
  }
  __syncthreads();

  // ---- criss-cross attention: 1024 (token,k) items, 4 / thread ----
#pragma unroll
  for (int it = 0; it < 4; it++) {
    int item = tid + it * 256;
    int t = item >> 6, k = item & 63;
    float qv[8], kk[8], vv[8], cacc[8];
#pragma unroll
    for (int c = 0; c < 8; c++) {
      qv[c] = (float)qkv16[t][c][k];
      kk[c] = (float)qkv16[t][c][64 + k];
      vv[c] = (float)qkv16[t][c][128 + k];
    }
    cc_attn8(qv, kk, vv, cacc);
#pragma unroll
    for (int m = 0; m < 8; m++) sa16[t][m][k] = (_Float16)cacc[m];
  }
  __syncthreads();

  // ---- fc_s GEMM: [128,64] x [64,64], wave == row-tile ----
  {
    int mt = wave;
    const _Float16* Arow = &sa16[0][0][0] + (size_t)(mt * 16 + (lane & 15)) * KD;
    v16h a0 = load16(Arow, 0, lane), a1 = load16(Arow, 32, lane);
#pragma unroll
    for (int nt = 0; nt < 4; nt++) {
      int n = nt * 16 + (lane & 15);
      const _Float16* Brow = wT_fcs + (size_t)n * KD;
      v16h b0 = load16(Brow, 0, lane), b1 = load16(Brow, 32, lane);
      v8f acc = {};
      acc = wmma16(a0, b0, acc);
      acc = wmma16(a1, b1, acc);
      float bias = b_fcs[n];
      int mhi = (lane & 16) ? 8 : 0;
#pragma unroll
      for (int r = 0; r < 8; r++) {
        int R = mt * 16 + r + mhi;
        int t = R >> 3, m = R & 7;
        size_t gi = (size_t)(tok0 + t) * HD + m * KD + n;
        xin2[gi] = acc[r] + bias + xin_f32[gi];
      }
    }
  }
}

// ---------------------------------------------------------------------------
// Cross-attention block: LN2 -> q/kv WMMA -> attn -> fc_c WMMA -> x_ca, LN3 -> y16
// 64 KB LDS, phase-aliased.
// ---------------------------------------------------------------------------
__global__ void __launch_bounds__(256) cross_attn_kernel(
    const float* __restrict__ xin2, const float* __restrict__ ctxp,
    const float* __restrict__ ln2w, const float* __restrict__ ln2b,
    const _Float16* __restrict__ wT_q, const float* __restrict__ b_q,
    const _Float16* __restrict__ wT_kv, const float* __restrict__ b_kv,
    const _Float16* __restrict__ wT_fcc, const float* __restrict__ b_fcc,
    const float* __restrict__ ln3w, const float* __restrict__ ln3b,
    float* __restrict__ xca, _Float16* __restrict__ y16) {
  __shared__ alignas(16) char smem[65536];
  typedef _Float16 Q16T[CCH][KD];     // [16][8][64]  16 KB
  typedef _Float16 KV16T[CCH][128];   // [16][8][128] 32 KB
  typedef _Float16 H16T[HD];          // [16][512]    16 KB
  typedef float    XCAT[HD];          // [16][512]    32 KB
  Q16T*  q16  = reinterpret_cast<Q16T*>(smem);            // [0, 16K)
  KV16T* kv16 = reinterpret_cast<KV16T*>(smem + 16384);   // [16K, 48K)
  H16T*  h16  = reinterpret_cast<H16T*>(smem + 49152);    // [48K, 64K)
  Q16T*  ca16 = reinterpret_cast<Q16T*>(smem + 49152);    // aliases h16
  XCAT*  xcs  = reinterpret_cast<XCAT*>(smem);            // [0, 32K) after attn

  int tid = threadIdx.x, wave = tid >> 5, lane = tid & 31;
  int tok0 = blockIdx.x * 16;

  // (a) LN2 -> h16 (16 threads / token)
  {
    int tl = tid >> 4, sub = tid & 15;
    const float* row = xin2 + (size_t)(tok0 + tl) * HD;
    float v[32]; float s = 0.f;
#pragma unroll
    for (int i = 0; i < 32; i++) { v[i] = row[sub + 16 * i]; s += v[i]; }
#pragma unroll
    for (int off = 8; off > 0; off >>= 1) s += __shfl_xor(s, off, 16);
    float mu = s * (1.f / 512.f);
    float var = 0.f;
#pragma unroll
    for (int i = 0; i < 32; i++) { float d = v[i] - mu; var += d * d; }
#pragma unroll
    for (int off = 8; off > 0; off >>= 1) var += __shfl_xor(var, off, 16);
    float rstd = rsqrtf(var * (1.f / 512.f) + 1e-6f);
#pragma unroll
    for (int i = 0; i < 32; i++) {
      int e = sub + 16 * i;
      h16[tl][e] = (_Float16)((v[i] - mu) * rstd * ln2w[e] + ln2b[e]);
    }
  }
  __syncthreads();

  // (b) q GEMM: wave == channel
  {
    int c = wave;
    const _Float16* Arow = &h16[lane & 15][c * KD];
    v16h a0 = load16(Arow, 0, lane), a1 = load16(Arow, 32, lane);
#pragma unroll
    for (int nt = 0; nt < 4; nt++) {
      int n = nt * 16 + (lane & 15);
      const _Float16* Brow = wT_q + ((size_t)c * KD + n) * KD;
      v16h b0 = load16(Brow, 0, lane), b1 = load16(Brow, 32, lane);
      v8f acc = {};
      acc = wmma16(a0, b0, acc);
      acc = wmma16(a1, b1, acc);
      float bias = b_q[c * KD + n];
      int mhi = (lane & 16) ? 8 : 0;
#pragma unroll
      for (int r = 0; r < 8; r++) q16[r + mhi][c][n] = (_Float16)(acc[r] + bias);
    }
  }
  __syncthreads();

  // (c) context -> f16, reuse h16 slot
#pragma unroll
  for (int p = 0; p < 32; p++) {
    int i = tid + p * 256;
    int tl = i >> 9, e = i & 511;
    h16[tl][e] = (_Float16)ctxp[(size_t)(tok0 + tl) * HD + e];
  }
  __syncthreads();

  // (d) kv GEMM: [16,64] x [64,128] per channel
  {
    int c = wave;
    const _Float16* Arow = &h16[lane & 15][c * KD];
    v16h a0 = load16(Arow, 0, lane), a1 = load16(Arow, 32, lane);
#pragma unroll
    for (int nt = 0; nt < 8; nt++) {
      int n = nt * 16 + (lane & 15);
      const _Float16* Brow = wT_kv + ((size_t)c * 128 + n) * KD;
      v16h b0 = load16(Brow, 0, lane), b1 = load16(Brow, 32, lane);
      v8f acc = {};
      acc = wmma16(a0, b0, acc);
      acc = wmma16(a1, b1, acc);
      float bias = b_kv[c * 128 + n];
      int mhi = (lane & 16) ? 8 : 0;
#pragma unroll
      for (int r = 0; r < 8; r++) kv16[r + mhi][c][n] = (_Float16)(acc[r] + bias);
    }
  }
  __syncthreads();

  // (e) criss-cross attention -> ca16 (aliases h16, dead now)
#pragma unroll
  for (int it = 0; it < 4; it++) {
    int item = tid + it * 256;
    int t = item >> 6, k = item & 63;
    float qv[8], kk[8], vv[8], cacc[8];
#pragma unroll
    for (int c = 0; c < 8; c++) {
      qv[c] = (float)q16[t][c][k];
      kk[c] = (float)kv16[t][c][k];
      vv[c] = (float)kv16[t][c][64 + k];
    }
    cc_attn8(qv, kk, vv, cacc);
#pragma unroll
    for (int m = 0; m < 8; m++) ca16[t][m][k] = (_Float16)cacc[m];
  }
  __syncthreads();

  // (f) fc_c GEMM -> x_ca (LDS f32 + global f32)
  {
    int mt = wave;
    const _Float16* Arow = &ca16[0][0][0] + (size_t)(mt * 16 + (lane & 15)) * KD;
    v16h a0 = load16(Arow, 0, lane), a1 = load16(Arow, 32, lane);
#pragma unroll
    for (int nt = 0; nt < 4; nt++) {
      int n = nt * 16 + (lane & 15);
      const _Float16* Brow = wT_fcc + (size_t)n * KD;
      v16h b0 = load16(Brow, 0, lane), b1 = load16(Brow, 32, lane);
      v8f acc = {};
      acc = wmma16(a0, b0, acc);
      acc = wmma16(a1, b1, acc);
      float bias = b_fcc[n];
      int mhi = (lane & 16) ? 8 : 0;
#pragma unroll
      for (int r = 0; r < 8; r++) {
        int R = mt * 16 + r + mhi;
        int t = R >> 3, m = R & 7;
        float val = acc[r] + bias;
        xcs[t][m * KD + n] = val;
        xca[(size_t)(tok0 + t) * HD + m * KD + n] = val;
      }
    }
  }
  __syncthreads();

  // (g) LN3 from LDS -> y16
  {
    int tl = tid >> 4, sub = tid & 15;
    float v[32]; float s = 0.f;
#pragma unroll
    for (int i = 0; i < 32; i++) { v[i] = xcs[tl][sub + 16 * i]; s += v[i]; }
#pragma unroll
    for (int off = 8; off > 0; off >>= 1) s += __shfl_xor(s, off, 16);
    float mu = s * (1.f / 512.f);
    float var = 0.f;
#pragma unroll
    for (int i = 0; i < 32; i++) { float d = v[i] - mu; var += d * d; }
#pragma unroll
    for (int off = 8; off > 0; off >>= 1) var += __shfl_xor(var, off, 16);
    float rstd = rsqrtf(var * (1.f / 512.f) + 1e-6f);
#pragma unroll
    for (int i = 0; i < 32; i++) {
      int e = sub + 16 * i;
      y16[(size_t)(tok0 + tl) * HD + e] = (_Float16)((v[i] - mu) * rstd * ln3w[e] + ln3b[e]);
    }
  }
}

// ---------------------------------------------------------------------------
// MLP GEMM1: [NT,512]x[512,2048] + b1, exact GELU -> t16
// block tile 64x128; wave == N-tile (16 cols, no redundant B loads),
// loops 4 M-tiles with A fragments from LDS.
// ---------------------------------------------------------------------------
__global__ void __launch_bounds__(256) mlp1_kernel(
    const _Float16* __restrict__ y16, const _Float16* __restrict__ wT1,
    const float* __restrict__ b1, _Float16* __restrict__ t16) {
  __shared__ alignas(16) _Float16 As[64][80];
  int tid = threadIdx.x, wave = tid >> 5, lane = tid & 31;
  int row0 = blockIdx.x * 64;
  int col0 = blockIdx.y * 128;
  int n = col0 + wave * 16 + (lane & 15);      // this wave's output column
  const _Float16* BrowBase = wT1 + (size_t)n * HD;
  v8f zero = {};
  v8f acc[4] = {zero, zero, zero, zero};       // one per M-tile
  for (int kc = 0; kc < 8; kc++) {
    __syncthreads();
#pragma unroll
    for (int p = 0; p < 2; p++) {
      int i = tid + p * 256;
      int r = i >> 3, c8 = i & 7;
      *(int4*)(&As[r][c8 * 8]) =
          *(const int4*)(y16 + (size_t)(row0 + r) * HD + kc * 64 + c8 * 8);
    }
    __syncthreads();
    const _Float16* Brow = BrowBase + kc * 64;
    if (kc + 1 < 8) __builtin_prefetch(Brow + 64, 0, 1);   // next B chunk
    v16h b0 = load16(Brow, 0, lane), b1v = load16(Brow, 32, lane);
#pragma unroll
    for (int mt = 0; mt < 4; mt++) {
      const _Float16* Arow = &As[mt * 16 + (lane & 15)][0];
      v16h a0 = load16(Arow, 0, lane), a1 = load16(Arow, 32, lane);
      acc[mt] = wmma16(a0, b0, acc[mt]);
      acc[mt] = wmma16(a1, b1v, acc[mt]);
    }
  }
  int mhi = (lane & 16) ? 8 : 0;
  float bias = b1[n];
#pragma unroll
  for (int mt = 0; mt < 4; mt++) {
#pragma unroll
    for (int r = 0; r < 8; r++) {
      float v = acc[mt][r] + bias;
      v = 0.5f * v * (1.0f + erff(v * 0.70710678118654752f));
      t16[(size_t)(row0 + mt * 16 + r + mhi) * MD + n] = (_Float16)v;
    }
  }
}

// ---------------------------------------------------------------------------
// MLP GEMM2: [NT,2048]x[2048,512] + b2 + x_ca -> out (f32)
// ---------------------------------------------------------------------------
__global__ void __launch_bounds__(256) mlp2_kernel(
    const _Float16* __restrict__ t16, const _Float16* __restrict__ wT2,
    const float* __restrict__ b2, const float* __restrict__ xca,
    float* __restrict__ out) {
  __shared__ alignas(16) _Float16 As[64][80];
  int tid = threadIdx.x, wave = tid >> 5, lane = tid & 31;
  int row0 = blockIdx.x * 64;
  int col0 = blockIdx.y * 128;
  int n = col0 + wave * 16 + (lane & 15);
  const _Float16* BrowBase = wT2 + (size_t)n * MD;
  v8f zero = {};
  v8f acc[4] = {zero, zero, zero, zero};
  for (int kc = 0; kc < 32; kc++) {
    __syncthreads();
#pragma unroll
    for (int p = 0; p < 2; p++) {
      int i = tid + p * 256;
      int r = i >> 3, c8 = i & 7;
      *(int4*)(&As[r][c8 * 8]) =
          *(const int4*)(t16 + (size_t)(row0 + r) * MD + kc * 64 + c8 * 8);
    }
    __syncthreads();
    const _Float16* Brow = BrowBase + kc * 64;
    if (kc + 1 < 32) __builtin_prefetch(Brow + 64, 0, 1);  // next B chunk
    v16h b0 = load16(Brow, 0, lane), b1v = load16(Brow, 32, lane);
#pragma unroll
    for (int mt = 0; mt < 4; mt++) {
      const _Float16* Arow = &As[mt * 16 + (lane & 15)][0];
      v16h a0 = load16(Arow, 0, lane), a1 = load16(Arow, 32, lane);
      acc[mt] = wmma16(a0, b0, acc[mt]);
      acc[mt] = wmma16(a1, b1v, acc[mt]);
    }
  }
  int mhi = (lane & 16) ? 8 : 0;
  float bias = b2[n];
#pragma unroll
  for (int mt = 0; mt < 4; mt++) {
#pragma unroll
    for (int r = 0; r < 8; r++) {
      size_t gi = (size_t)(row0 + mt * 16 + r + mhi) * HD + n;
      out[gi] = acc[mt][r] + bias + xca[gi];
    }
  }
}

// ---------------------------------------------------------------------------
extern "C" void kernel_launch(void* const* d_in, const int* in_sizes, int n_in,
                              void* d_out, int out_size, void* d_ws, size_t ws_size,
                              hipStream_t stream) {
  (void)in_sizes; (void)n_in; (void)out_size; (void)ws_size;
  const float* x      = (const float*)d_in[0];
  const float* ctxp   = (const float*)d_in[1];
  const float* ln1_w  = (const float*)d_in[2];
  const float* ln1_b  = (const float*)d_in[3];
  const float* w_qkv  = (const float*)d_in[4];
  const float* b_qkv  = (const float*)d_in[5];
  const float* w_fc_s = (const float*)d_in[6];
  const float* b_fc_s = (const float*)d_in[7];
  const float* ln2_w  = (const float*)d_in[8];
  const float* ln2_b  = (const float*)d_in[9];
  const float* w_q    = (const float*)d_in[10];
  const float* b_q    = (const float*)d_in[11];
  const float* w_kv   = (const float*)d_in[12];
  const float* b_kv   = (const float*)d_in[13];
  const float* w_fc_c = (const float*)d_in[14];
  const float* b_fc_c = (const float*)d_in[15];
  const float* ln3_w  = (const float*)d_in[16];
  const float* ln3_b  = (const float*)d_in[17];
  const float* w1     = (const float*)d_in[18];
  const float* b1     = (const float*)d_in[19];
  const float* w2     = (const float*)d_in[20];
  const float* b2     = (const float*)d_in[21];

  char* ws = (char*)d_ws;
  size_t off = 0;
  auto alloc = [&](size_t bytes) -> char* {
    char* p = ws + off;
    off += (bytes + 255) & ~(size_t)255;
    return p;
  };
  float*    xin    = (float*)alloc((size_t)NTOK * HD * 4);
  _Float16* xin16  = (_Float16*)alloc((size_t)NTOK * HD * 2);
  float*    xin2   = (float*)alloc((size_t)NTOK * HD * 4);
  float*    xca    = (float*)alloc((size_t)NTOK * HD * 4);
  _Float16* y16    = (_Float16*)alloc((size_t)NTOK * HD * 2);
  _Float16* t16    = (_Float16*)alloc((size_t)NTOK * MD * 2);
  _Float16* wT_qkv = (_Float16*)alloc((size_t)CCH * 192 * KD * 2);
  _Float16* wT_fcs = (_Float16*)alloc((size_t)KD * KD * 2);
  _Float16* wT_q   = (_Float16*)alloc((size_t)CCH * KD * KD * 2);
  _Float16* wT_kv  = (_Float16*)alloc((size_t)CCH * 128 * KD * 2);
  _Float16* wT_fcc = (_Float16*)alloc((size_t)KD * KD * 2);
  _Float16* wT1    = (_Float16*)alloc((size_t)MD * HD * 2);
  _Float16* wT2    = (_Float16*)alloc((size_t)HD * MD * 2);

  auto tl = [&](const float* src, _Float16* dst, int B, int R, int N) {
    int total = B * R * N;
    transpose_f16_kernel<<<(total + 255) / 256, 256, 0, stream>>>(src, dst, B, R, N);
  };
  tl(w_qkv, wT_qkv, CCH, KD, 192);
  tl(w_fc_s, wT_fcs, 1, KD, KD);
  tl(w_q, wT_q, CCH, KD, KD);
  tl(w_kv, wT_kv, CCH, KD, 128);
  tl(w_fc_c, wT_fcc, 1, KD, KD);
  tl(w1, wT1, 1, HD, MD);
  tl(w2, wT2, 1, MD, HD);

  ln1_kernel<<<NTOK / 8, 256, 0, stream>>>(x, ln1_w, ln1_b, xin, xin16);
  self_attn_kernel<<<NTOK / 16, 256, 0, stream>>>(xin16, xin, wT_qkv, b_qkv,
                                                  wT_fcs, b_fc_s, xin2);
  cross_attn_kernel<<<NTOK / 16, 256, 0, stream>>>(xin2, ctxp, ln2_w, ln2_b,
                                                   wT_q, b_q, wT_kv, b_kv,
                                                   wT_fcc, b_fc_c, ln3_w, ln3_b,
                                                   xca, y16);
  mlp1_kernel<<<dim3(NTOK / 64, MD / 128), 256, 0, stream>>>(y16, wT1, b1, t16);
  mlp2_kernel<<<dim3(NTOK / 64, HD / 128), 256, 0, stream>>>(t16, wT2, b2, xca,
                                                             (float*)d_out);
}
